// TPAttention_88699664597439
// MI455X (gfx1250) — matrix operand order repaired and checked by reading, not compiled, not run on hardware
//
#include <hip/hip_runtime.h>
#include <hip/hip_bf16.h>

// ---------------- problem constants (match reference) ----------------
#define HIDDEN   2048
#define HEADS    16
#define HEAD_DIM 128           // HIDDEN / HEADS
#define BATCH    2
#define SEQ      2048
#define MROWS    (BATCH * SEQ) // 4096
#define QKV_N    (3 * HIDDEN)  // 6144
#define SCALE    0.08838834764831845f  // 1/sqrt(128)

typedef __attribute__((ext_vector_type(16))) _Float16 v16h;
typedef __attribute__((ext_vector_type(8)))  float    v8f;

// ---- CDNA5 async global->LDS copy (guarded; falls back to plain ld/st) ----
#if defined(__has_builtin)
#if __has_builtin(__builtin_amdgcn_global_load_async_to_lds_b64)
#define HAVE_ASYNC_LDS 1
#endif
#if __has_builtin(__builtin_amdgcn_s_wait_asynccnt)
#define HAVE_WAIT_ASYNC_BUILTIN 1
#endif
#endif

// data payload type of the b64 async copy: 2 x i32 vector
typedef __attribute__((__vector_size__(2 * sizeof(int)))) int async_b64_t;
#if HAVE_ASYNC_LDS
typedef __attribute__((address_space(1))) async_b64_t as1_b64_t;  // global
typedef __attribute__((address_space(3))) async_b64_t as3_b64_t;  // LDS
#endif

__device__ __forceinline__ void copy8B_to_lds(const _Float16* g, _Float16* l) {
#if HAVE_ASYNC_LDS
  __builtin_amdgcn_global_load_async_to_lds_b64((as1_b64_t*)g, (as3_b64_t*)l, 0, 0);
#else
  *(unsigned long long*)l = *(const unsigned long long*)g;
#endif
}

__device__ __forceinline__ void wait_async0() {
#if HAVE_ASYNC_LDS
#if HAVE_WAIT_ASYNC_BUILTIN
  __builtin_amdgcn_s_wait_asynccnt(0);
#else
  asm volatile("s_wait_asynccnt 0x0" ::: "memory");
#endif
#endif
}

// D = A(16x32 f16) * B(32x16 f16) + C(16x16 f32)
__device__ __forceinline__ v8f wmma16(v16h a, v16h b, v8f c) {
  return __builtin_amdgcn_wmma_f32_16x16x32_f16(
      false, a, false, b, (short)0, c, false, false);
}

__device__ __forceinline__ v8f vzero() {
  v8f z = {0.f, 0.f, 0.f, 0.f, 0.f, 0.f, 0.f, 0.f};
  return z;
}

// ---------------- fp32 -> fp16 convert ----------------
__global__ void k_cvt_f32_f16(const float* __restrict__ src,
                              _Float16* __restrict__ dst, size_t n) {
  size_t i = (size_t)blockIdx.x * blockDim.x + threadIdx.x;
  if (i < n) dst[i] = (_Float16)src[i];
}

// ---------------- interleaved-pair RoPE, in place on f16 QKV ----------------
__global__ void k_rope(_Float16* __restrict__ qkvh, const int* __restrict__ positions) {
  size_t idx = (size_t)blockIdx.x * blockDim.x + threadIdx.x;  // one pair / thread
  const int PAIRS = HEAD_DIM / 2;                               // 64
  if (idx >= (size_t)BATCH * SEQ * 2 * HEADS * PAIRS) return;
  int i = (int)(idx % PAIRS); size_t t = idx / PAIRS;
  int h = (int)(t % HEADS);   t /= HEADS;
  int part = (int)(t % 2);    t /= 2;
  int s = (int)(t % SEQ);     int b = (int)(t / SEQ);

  float pos = (float)positions[s];
  float inv = __expf(-((float)(2 * i) * (1.0f / HEAD_DIM)) * 9.2103403719761836f);
  float sn, cs; __sincosf(pos * inv, &sn, &cs);

  size_t off = ((size_t)(b * SEQ + s)) * QKV_N + (size_t)part * HIDDEN
             + (size_t)h * HEAD_DIM + 2 * i;
  float x0 = (float)qkvh[off], x1 = (float)qkvh[off + 1];
  qkvh[off]     = (_Float16)(x0 * cs - x1 * sn);
  qkvh[off + 1] = (_Float16)(x0 * sn + x1 * cs);
}

// ---------------- GEMM v2: C[m,n] = sum_k A[m,k] * W[n,k] ----------------
// Block 256 thr = 8 waves. Block tile 32(M) x 512(N); wave: 2 Afrag x 4 Bfrag.
// Shared 32x32 A tile double-buffered in LDS via async global->LDS copy.
#define APAD 8
template <typename OutT>
__global__ __launch_bounds__(256) void k_gemm_v2(const _Float16* __restrict__ A,
                                                 const _Float16* __restrict__ W,
                                                 OutT* __restrict__ C,
                                                 int N, int K) {
  const int tid = threadIdx.x, lane = tid & 31, wave = tid >> 5;
  const int ml = lane & 15, hi = lane >> 4;
  const int m0 = blockIdx.y * 32;
  const int n0 = blockIdx.x * 512 + wave * 64;

  __shared__ _Float16 At[2][32][32 + APAD];  // 2 x 2.5KB

  // cooperative copy mapping: 256 threads x 8B = 2KB tile
  const int crow = tid >> 3, cchunk = (tid & 7) * 4;
  const _Float16* gsrc = A + (size_t)(m0 + crow) * K + cchunk;

  copy8B_to_lds(gsrc, &At[0][crow][cchunk]);
  wait_async0();
  __syncthreads();

  v8f acc[2][4];
#pragma unroll
  for (int mf = 0; mf < 2; ++mf)
#pragma unroll
    for (int nf = 0; nf < 4; ++nf) acc[mf][nf] = vzero();

  for (int k0 = 0; k0 < K; k0 += 32) {
    const int cur = (k0 >> 5) & 1;
    if (k0 + 32 < K) copy8B_to_lds(gsrc + k0 + 32, &At[cur ^ 1][crow][cchunk]);

    v16h afr[2];
#pragma unroll
    for (int mf = 0; mf < 2; ++mf) {
      const _Float16* ap = &At[cur][mf * 16 + ml][hi * 16];
#pragma unroll
      for (int j = 0; j < 8; ++j) { afr[mf][j] = ap[j]; afr[mf][j + 8] = ap[8 + j]; }
    }
#pragma unroll
    for (int nf = 0; nf < 4; ++nf) {
      const _Float16* wr = W + (size_t)(n0 + nf * 16 + ml) * K + k0 + hi * 16;
      __builtin_prefetch(wr + 64, 0, 0);
      v16h bf;
#pragma unroll
      for (int j = 0; j < 16; ++j) bf[j] = wr[j];
      acc[0][nf] = wmma16(afr[0], bf, acc[0][nf]);
      acc[1][nf] = wmma16(afr[1], bf, acc[1][nf]);
    }
    wait_async0();
    __syncthreads();
  }

#pragma unroll
  for (int mf = 0; mf < 2; ++mf)
#pragma unroll
    for (int nf = 0; nf < 4; ++nf) {
      OutT* crow_p = C + (size_t)(m0 + mf * 16 + hi * 8) * N + n0 + nf * 16 + ml;
#pragma unroll
      for (int r = 0; r < 8; ++r) crow_p[(size_t)r * N] = (OutT)acc[mf][nf][r];
    }
}

// ---------------- fused attention: scores -> softmax -> P (out) -> P@V ----------------
// One block per (b, h, 16-query strip). The full 16x2048 score strip lives in
// registers (16 v8f accs per wave). P is written once to d_out (f32) and staged
// as f16 in LDS (two 1024-col halves) for the P@V WMMA A-fragments.
#define PPAD 8
__global__ __launch_bounds__(256) void k_attn_fused(const _Float16* __restrict__ qkvh,
                                                    float* __restrict__ attnW,
                                                    _Float16* __restrict__ aouth) {
  const int tid = threadIdx.x, lane = tid & 31, wave = tid >> 5;
  const int ml = lane & 15, hi = lane >> 4;
  const int bh = blockIdx.y, b = bh >> 4, h = bh & 15;
  const int q0 = blockIdx.x * 16;

  __shared__ _Float16 Pl[16][1024 + PPAD];  // 33 KB
  __shared__ float red[16][8];

  const _Float16* qrow = qkvh + ((size_t)b * SEQ + q0 + ml) * QKV_N
                       + (size_t)h * HEAD_DIM + hi * 16;
  const _Float16* kbase = qkvh + ((size_t)b * SEQ) * QKV_N + HIDDEN
                        + (size_t)h * HEAD_DIM;

  // ---- phase 1: scores. wave covers key cols [wave*256, wave*256+256) ----
  v8f acc[16];
#pragma unroll
  for (int f = 0; f < 16; ++f) acc[f] = vzero();

#pragma unroll
  for (int d0 = 0; d0 < HEAD_DIM; d0 += 32) {
    v16h a;
#pragma unroll
    for (int j = 0; j < 8; ++j) { a[j] = qrow[d0 + j]; a[j + 8] = qrow[d0 + 8 + j]; }
#pragma unroll
    for (int f = 0; f < 16; ++f) {
      const _Float16* kr = kbase + (size_t)(wave * 256 + f * 16 + ml) * QKV_N
                         + d0 + hi * 16;
      v16h bf;
#pragma unroll
      for (int j = 0; j < 16; ++j) bf[j] = kr[j];
      acc[f] = wmma16(a, bf, acc[f]);
    }
  }

  // ---- phase 2: row softmax. thread holds rows hi*8+r, cols wave*256+f*16+ml ----
  float rmax[8], rinv[8];
#pragma unroll
  for (int r = 0; r < 8; ++r) {
    float m = -3.402823466e38f;
#pragma unroll
    for (int f = 0; f < 16; ++f) { acc[f][r] *= SCALE; m = fmaxf(m, acc[f][r]); }
#pragma unroll
    for (int off = 8; off; off >>= 1) m = fmaxf(m, __shfl_xor(m, off));
    if (ml == 0) red[hi * 8 + r][wave] = m;
  }
  __syncthreads();
#pragma unroll
  for (int r = 0; r < 8; ++r) {
    float m = red[hi * 8 + r][0];
#pragma unroll
    for (int w = 1; w < 8; ++w) m = fmaxf(m, red[hi * 8 + r][w]);
    rmax[r] = m;
  }
  __syncthreads();
#pragma unroll
  for (int r = 0; r < 8; ++r) {
    float s = 0.f;
#pragma unroll
    for (int f = 0; f < 16; ++f) { acc[f][r] = __expf(acc[f][r] - rmax[r]); s += acc[f][r]; }
#pragma unroll
    for (int off = 8; off; off >>= 1) s += __shfl_xor(s, off);
    if (ml == 0) red[hi * 8 + r][wave] = s;
  }
  __syncthreads();
#pragma unroll
  for (int r = 0; r < 8; ++r) {
    float s = 0.f;
#pragma unroll
    for (int w = 0; w < 8; ++w) s += red[hi * 8 + r][w];
    rinv[r] = 1.0f / s;
  }

  // ---- phase 3: write normalized P to d_out (f32), once ----
  float* pg = attnW + ((size_t)bh * SEQ + q0) * SEQ;
#pragma unroll
  for (int f = 0; f < 16; ++f) {
    const int col = wave * 256 + f * 16 + ml;
#pragma unroll
    for (int r = 0; r < 8; ++r)
      pg[(size_t)(hi * 8 + r) * SEQ + col] = acc[f][r] * rinv[r];
  }

  // ---- phase 4: P@V in two 1024-col halves via LDS-staged f16 P ----
  const _Float16* vbase = qkvh + ((size_t)b * SEQ) * QKV_N + 2 * HIDDEN
                        + (size_t)h * HEAD_DIM;
  const int n0 = wave * 16;  // 8 waves cover HEAD_DIM = 128
  v8f oacc = vzero();

#pragma unroll
  for (int half = 0; half < 2; ++half) {
    __syncthreads();
    if ((wave >> 2) == half) {
      const int colbase = wave * 256 - half * 1024;
#pragma unroll
      for (int f = 0; f < 16; ++f)
#pragma unroll
        for (int r = 0; r < 8; ++r)
          Pl[hi * 8 + r][colbase + f * 16 + ml] = (_Float16)(acc[f][r] * rinv[r]);
    }
    __syncthreads();
    for (int k0 = 0; k0 < 1024; k0 += 32) {
      v16h a;
      const _Float16* pr = &Pl[ml][k0 + hi * 16];
#pragma unroll
      for (int j = 0; j < 8; ++j) { a[j] = pr[j]; a[j + 8] = pr[8 + j]; }
      v16h bf;
      const _Float16* vr = vbase + (size_t)(half * 1024 + k0 + hi * 16) * QKV_N + n0 + ml;
#pragma unroll
      for (int j = 0; j < 16; ++j) bf[j] = vr[(size_t)j * QKV_N];
      oacc = wmma16(a, bf, oacc);
    }
  }

  _Float16* orow = aouth + ((size_t)(b * SEQ) + q0 + hi * 8) * HIDDEN
                 + (size_t)h * HEAD_DIM + n0 + ml;
#pragma unroll
  for (int r = 0; r < 8; ++r) orow[(size_t)r * HIDDEN] = (_Float16)oacc[r];
}

// ---------------- host launch ----------------
extern "C" void kernel_launch(void* const* d_in, const int* in_sizes, int n_in,
                              void* d_out, int out_size, void* d_ws, size_t ws_size,
                              hipStream_t stream) {
  const float* x     = (const float*)d_in[0];   // [B,S,HIDDEN]
  const float* w_qkv = (const float*)d_in[1];   // [3H,H]
  const float* w_o   = (const float*)d_in[2];   // [H,H]
  const int*   pos   = (const int*)d_in[3];     // [S]

  float* out   = (float*)d_out;                        // [B,S,HIDDEN]
  float* attnW = out + (size_t)BATCH * SEQ * HIDDEN;   // [B,H,S,S]

  char* w = (char*)d_ws;
  _Float16* xh    = (_Float16*)w; w += (size_t)MROWS * HIDDEN * 2;   // 16 MiB
  _Float16* wqkvh = (_Float16*)w; w += (size_t)QKV_N * HIDDEN * 2;   // 24 MiB
  _Float16* woh   = (_Float16*)w; w += (size_t)HIDDEN * HIDDEN * 2;  //  8 MiB
  _Float16* qkvh  = (_Float16*)w; w += (size_t)MROWS * QKV_N * 2;    // 48 MiB
  _Float16* aouth = (_Float16*)w; w += (size_t)MROWS * HIDDEN * 2;   // 16 MiB

  const size_t nx = (size_t)MROWS * HIDDEN;
  const size_t nq = (size_t)QKV_N * HIDDEN;
  const size_t nw = (size_t)HIDDEN * HIDDEN;
  k_cvt_f32_f16<<<dim3((nx + 255) / 256), 256, 0, stream>>>(x,     xh,    nx);
  k_cvt_f32_f16<<<dim3((nq + 255) / 256), 256, 0, stream>>>(w_qkv, wqkvh, nq);
  k_cvt_f32_f16<<<dim3((nw + 255) / 256), 256, 0, stream>>>(w_o,   woh,   nw);

  // QKV projection: [4096,2048] @ [6144,2048]^T -> f16 [4096,6144]
  k_gemm_v2<_Float16><<<dim3(QKV_N / 512, MROWS / 32), 256, 0, stream>>>(
      xh, wqkvh, qkvh, QKV_N, HIDDEN);

  // RoPE in place on Q and K
  const size_t npairs = (size_t)BATCH * SEQ * 2 * HEADS * (HEAD_DIM / 2);
  k_rope<<<dim3((npairs + 255) / 256), 256, 0, stream>>>(qkvh, pos);

  // fused scores + softmax + attn_weights write + P@V
  k_attn_fused<<<dim3(SEQ / 16, BATCH * HEADS), 256, 0, stream>>>(qkvh, attnW, aouth);

  // output projection: [4096,2048] @ [2048,2048]^T -> f32 out
  k_gemm_v2<float><<<dim3(HIDDEN / 512, MROWS / 32), 256, 0, stream>>>(
      aouth, woh, out, HIDDEN, HIDDEN);
}
